// Model_73916387164836
// MI455X (gfx1250) — compile-verified, matrix operand order checked by reading, not run on hardware
//
#include <hip/hip_runtime.h>
#include <hip/hip_bf16.h>

typedef __attribute__((ext_vector_type(16))) __bf16 bf16x16;
typedef __attribute__((ext_vector_type(8)))  float  f32x8;

// Problem dims
constexpr int cB     = 512;
constexpr int cM     = 196;   // student local tokens
constexpr int cN     = 77;    // text tokens
constexpr int cFEAT  = 512;
constexpr int cNCLS  = 632;
constexpr int cTDIM  = 768;
constexpr int cLDIM  = 576;

// accumulator slots
#define ACC_CET   0
#define ACC_CES   1
#define ACC_KL    2
#define ACC_TRT   3
#define ACC_TRS   4
#define ACC_SIMOP 5

// ---------------------------------------------------------------------------
// WMMA TN GEMM:  C[M,N] = A[M,K] * B[N,K]^T + bias   (bf16 MACs, f32 accum)
// Fragment layout per CDNA5 ISA 7.12.2 (16-bit A 16x32):
//   lane L (0..15):  row M=L,    halves 0..7 -> K=k0+0..7,  halves 8..15 -> K=k0+16..23
//   lane L (16..31): row M=L-16, halves 0..7 -> K=k0+8..15, halves 8..15 -> K=k0+24..31
// B (32x16, K-major per column) uses the identical pattern with row -> column.
// Each wave computes a 32x32 C tile: 2 A-frags x 2 B-frags -> 4 wmma / K-step.
// Block = 8 waves = 128(M) x 64(N) of C.
// ---------------------------------------------------------------------------
template <typename T, bool EDGE>
__device__ inline bf16x16 load_frag_tn(const T* __restrict__ rowp, int k0, int lane, bool valid) {
    bf16x16 f;
    if (!EDGE || valid) {
        const int kb = (lane >> 4) << 3;      // 0 or 8
        const T* p0 = rowp + k0 + kb;         // 8 contiguous elems
        const T* p1 = rowp + k0 + 16 + kb;    // 8 contiguous elems
#pragma unroll
        for (int h = 0; h < 8; ++h) f[h]     = (__bf16)(float)p0[h];
#pragma unroll
        for (int h = 0; h < 8; ++h) f[8 + h] = (__bf16)(float)p1[h];
    } else {
#pragma unroll
        for (int h = 0; h < 16; ++h) f[h] = (__bf16)0.0f;
    }
    return f;
}

template <bool EDGE>
__device__ inline void store_tile(float* __restrict__ C, int M, int N,
                                  int tileM, int tileN, int lane,
                                  f32x8 acc, const float* __restrict__ bias) {
    const int r0 = tileM + ((lane >> 4) << 3);
    const int cc = tileN + (lane & 15);
    if (EDGE && cc >= N) return;
    const float bv = bias ? bias[cc] : 0.0f;
#pragma unroll
    for (int i = 0; i < 8; ++i) {
        int r = r0 + i;
        if (!EDGE || r < M) C[(long)r * N + cc] = acc[i] + bv;
    }
}

template <typename T, bool EDGE>
__global__ __launch_bounds__(256)
void gemm_tn_wmma(const T* __restrict__ A, const T* __restrict__ Bm,
                  const float* __restrict__ bias, float* __restrict__ C,
                  int M, int N, int K,
                  long strideA, long strideB, long strideC) {
    const int batch = blockIdx.z;
    A  += (long)batch * strideA;
    Bm += (long)batch * strideB;
    C  += (long)batch * strideC;

    const int wave = threadIdx.x >> 5;
    const int lane = threadIdx.x & 31;
    const int waveM = (blockIdx.y * 4 + (wave >> 1)) * 32;   // 32 rows per wave
    const int waveN = (blockIdx.x * 2 + (wave & 1)) * 32;    // 32 cols per wave
    if (waveM >= M || waveN >= N) return;                    // wave-uniform exit

    const int aR0 = waveM + (lane & 15), aR1 = aR0 + 16;
    const int bR0 = waveN + (lane & 15), bR1 = bR0 + 16;
    const bool aV0 = !EDGE || aR0 < M, aV1 = !EDGE || aR1 < M;
    const bool bV0 = !EDGE || bR0 < N, bV1 = !EDGE || bR1 < N;
    const T* a0 = A  + (long)aR0 * K;
    const T* a1 = A  + (long)aR1 * K;
    const T* b0 = Bm + (long)bR0 * K;
    const T* b1 = Bm + (long)bR1 * K;

    f32x8 acc00 = {0.f,0.f,0.f,0.f,0.f,0.f,0.f,0.f};
    f32x8 acc01 = acc00, acc10 = acc00, acc11 = acc00;

    for (int k0 = 0; k0 < K; k0 += 32) {
        // speculative prefetch of the A streams (B/weights are L2-resident);
        // OOB/unmapped prefetches are dropped by hardware.
        __builtin_prefetch(a0 + k0 + 96, 0, 1);
        __builtin_prefetch(a1 + k0 + 96, 0, 1);
        bf16x16 fa0 = load_frag_tn<T, EDGE>(a0, k0, lane, aV0);
        bf16x16 fa1 = load_frag_tn<T, EDGE>(a1, k0, lane, aV1);
        bf16x16 fb0 = load_frag_tn<T, EDGE>(b0, k0, lane, bV0);
        bf16x16 fb1 = load_frag_tn<T, EDGE>(b1, k0, lane, bV1);
        acc00 = __builtin_amdgcn_wmma_f32_16x16x32_bf16(false, fa0, false, fb0, (short)0, acc00, false, false);
        acc01 = __builtin_amdgcn_wmma_f32_16x16x32_bf16(false, fa0, false, fb1, (short)0, acc01, false, false);
        acc10 = __builtin_amdgcn_wmma_f32_16x16x32_bf16(false, fa1, false, fb0, (short)0, acc10, false, false);
        acc11 = __builtin_amdgcn_wmma_f32_16x16x32_bf16(false, fa1, false, fb1, (short)0, acc11, false, false);
    }

    store_tile<EDGE>(C, M, N, waveM,      waveN,      lane, acc00, bias);
    store_tile<EDGE>(C, M, N, waveM,      waveN + 16, lane, acc01, bias);
    store_tile<EDGE>(C, M, N, waveM + 16, waveN,      lane, acc10, bias);
    store_tile<EDGE>(C, M, N, waveM + 16, waveN + 16, lane, acc11, bias);
}

// ---------------------------------------------------------------------------
// BatchNorm (training stats over batch dim), in place.  One block per column.
// ---------------------------------------------------------------------------
__global__ __launch_bounds__(256)
void bn_train_kernel(float* __restrict__ x, const float* __restrict__ g,
                     const float* __restrict__ b, int rows, int cols) {
    const int j = blockIdx.x, tid = threadIdx.x;
    __shared__ float s1[256], s2[256];
    float sum = 0.f, sq = 0.f;
    for (int i = tid; i < rows; i += 256) {
        float v = x[(long)i * cols + j];
        sum += v; sq += v * v;
    }
    s1[tid] = sum; s2[tid] = sq; __syncthreads();
    for (int k = 128; k > 0; k >>= 1) {
        if (tid < k) { s1[tid] += s1[tid + k]; s2[tid] += s2[tid + k]; }
        __syncthreads();
    }
    const float mu  = s1[0] / rows;
    const float var = s2[0] / rows - mu * mu;
    const float sc  = g[j] * rsqrtf(var + 1e-5f);
    const float bb  = b[j];
    for (int i = tid; i < rows; i += 256) {
        long idx = (long)i * cols + j;
        x[idx] = (x[idx] - mu) * sc + bb;
    }
}

// ---------------------------------------------------------------------------
// L2 row-normalize (f32 -> bf16).  One block per row.
// ---------------------------------------------------------------------------
__global__ __launch_bounds__(256)
void rownorm_bf16_kernel(const float* __restrict__ in, __bf16* __restrict__ out, int D) {
    const long row = blockIdx.x;
    const float* p = in + row * D;
    __bf16* q = out + row * D;
    const int tid = threadIdx.x;
    __shared__ float s[256];
    float ss = 0.f;
    for (int i = tid; i < D; i += 256) { float v = p[i]; ss += v * v; }
    s[tid] = ss; __syncthreads();
    for (int k = 128; k > 0; k >>= 1) { if (tid < k) s[tid] += s[tid + k]; __syncthreads(); }
    const float inv = 1.0f / fmaxf(sqrtf(s[0]), 1e-12f);
    for (int i = tid; i < D; i += 256) q[i] = (__bf16)(p[i] * inv);
}

// ---------------------------------------------------------------------------
// Sinkhorn per batch: K = exp((sim-1)/0.1) resident in LDS (196x77 = 60 KB).
// After convergence accumulates sum(T * sim) into acc[ACC_SIMOP].
// ---------------------------------------------------------------------------
__global__ __launch_bounds__(256)
void sinkhorn_kernel(const float* __restrict__ sim, float* __restrict__ acc) {
    constexpr int SM = cM, SN = cN;
    const int b = blockIdx.x, tid = threadIdx.x;
    __shared__ float Ks[SM * SN];
    __shared__ float r[SM];
    __shared__ float c[SN];
    __shared__ float red[256];
    __shared__ float err;

    const float* simb = sim + (long)b * SM * SN;
    for (int i = tid; i < SM * SN; i += 256) Ks[i] = __expf((simb[i] - 1.0f) * 10.0f);
    if (tid < SM) r[tid] = 1.0f;
    if (tid < SN) c[tid] = 1.0f;
    __syncthreads();

    const float u = 1.0f / SM, v = 1.0f / SN;
    for (int it = 0; it < 100; ++it) {
        float d = 0.f;
        if (tid < SM) {
            float s = 0.f;
            for (int n = 0; n < SN; ++n) s += Ks[tid * SN + n] * c[n];
            float rn = u / s;
            d = fabsf(rn - r[tid]);
            r[tid] = rn;
        }
        red[tid] = d;
        __syncthreads();
        if (tid < SN) {
            float s = 0.f;
            for (int m = 0; m < SM; ++m) s += Ks[m * SN + tid] * r[m];
            c[tid] = v / s;
        }
        for (int k = 128; k > 0; k >>= 1) { if (tid < k) red[tid] += red[tid + k]; __syncthreads(); }
        if (tid == 0) err = red[0] / SM;
        __syncthreads();
        if (err < 0.01f) break;
    }

    float t = 0.f;
    for (int i = tid; i < SM * SN; i += 256) {
        int m = i / SN, n = i - m * SN;
        t += r[m] * c[n] * Ks[i] * simb[i];
    }
    red[tid] = t; __syncthreads();
    for (int k = 128; k > 0; k >>= 1) { if (tid < k) red[tid] += red[tid + k]; __syncthreads(); }
    if (tid == 0) atomicAdd(&acc[ACC_SIMOP], red[0]);
}

// ---------------------------------------------------------------------------
// CE(label-smooth) for both logits + KL distill (TEMP=2) + student argmax.
// One 256-thread block per batch row.
// ---------------------------------------------------------------------------
__device__ inline float block_sum256(float v, float* red) {
    const int tid = threadIdx.x;
    red[tid] = v; __syncthreads();
    for (int k = 128; k > 0; k >>= 1) { if (tid < k) red[tid] += red[tid + k]; __syncthreads(); }
    float r = red[0]; __syncthreads();
    return r;
}

__global__ __launch_bounds__(256)
void ce_kl_kernel(const float* __restrict__ lt, const float* __restrict__ ls,
                  const int* __restrict__ label, float* __restrict__ acc,
                  float* __restrict__ preds) {
    constexpr int C = cNCLS;
    const int row = blockIdx.x, tid = threadIdx.x;
    const float* t = lt + (long)row * C;
    const float* s = ls + (long)row * C;
    __shared__ float red[256];
    __shared__ int   redi[256];

    float mt = -__builtin_inff(), ms = -__builtin_inff();
    int ai = 0;
    for (int j = tid; j < C; j += 256) {
        float tv = t[j]; if (tv > mt) mt = tv;
        float sv = s[j]; if (sv > ms) { ms = sv; ai = j; }
    }
    red[tid] = mt; __syncthreads();
    for (int k = 128; k > 0; k >>= 1) { if (tid < k) red[tid] = fmaxf(red[tid], red[tid + k]); __syncthreads(); }
    mt = red[0]; __syncthreads();

    red[tid] = ms; redi[tid] = ai; __syncthreads();
    for (int k = 128; k > 0; k >>= 1) {
        if (tid < k) {
            if (red[tid + k] > red[tid] ||
                (red[tid + k] == red[tid] && redi[tid + k] < redi[tid])) {
                red[tid] = red[tid + k]; redi[tid] = redi[tid + k];
            }
        }
        __syncthreads();
    }
    ms = red[0];
    const int amax = redi[0];
    __syncthreads();

    float e_t = 0.f, e_s = 0.f, eT_t = 0.f, eT_s = 0.f, su_t = 0.f, su_s = 0.f;
    for (int j = tid; j < C; j += 256) {
        float tv = t[j], sv = s[j];
        e_t  += __expf(tv - mt);          e_s  += __expf(sv - ms);
        eT_t += __expf(0.5f * (tv - mt)); eT_s += __expf(0.5f * (sv - ms));
        su_t += tv;                        su_s += sv;
    }
    e_t  = block_sum256(e_t, red);   e_s  = block_sum256(e_s, red);
    eT_t = block_sum256(eT_t, red);  eT_s = block_sum256(eT_s, red);
    su_t = block_sum256(su_t, red);  su_s = block_sum256(su_s, red);

    const float lse_t  = logf(e_t),  lse_s  = logf(e_s);
    const float lseT_t = logf(eT_t), lseT_s = logf(eT_s);

    float kl = 0.f;
    for (int j = tid; j < C; j += 256) {
        float tl = 0.5f * (t[j] - mt) - lseT_t;   // log_softmax(teacher/2)
        float sl = 0.5f * (s[j] - ms) - lseT_s;   // log_softmax(student/2)
        kl += __expf(tl) * (tl - sl);
    }
    kl = block_sum256(kl, red);

    if (tid == 0) {
        const int lbl = label[row];
        const float lpt = t[lbl] - mt - lse_t;
        const float lps = s[lbl] - ms - lse_s;
        const float sum_logp_t = su_t - (float)C * (mt + lse_t);
        const float sum_logp_s = su_s - (float)C * (ms + lse_s);
        const float ce_t = -((1.0f - 0.1f) * lpt + (0.1f / C) * sum_logp_t);
        const float ce_s = -((1.0f - 0.1f) * lps + (0.1f / C) * sum_logp_s);
        atomicAdd(&acc[ACC_CET], ce_t);
        atomicAdd(&acc[ACC_CES], ce_s);
        atomicAdd(&acc[ACC_KL],  kl);
        preds[row] = (float)amax;
    }
}

// ---------------------------------------------------------------------------
// Triplet hard mining from precomputed Gram matrix G = feat @ feat^T.
// One block per anchor.
// ---------------------------------------------------------------------------
__global__ __launch_bounds__(256)
void triplet_kernel(const float* __restrict__ G, const int* __restrict__ label,
                    float* __restrict__ acc, int accIdx) {
    constexpr int Bn = cB;
    const int i = blockIdx.x, tid = threadIdx.x;
    __shared__ float red[256];
    const int   li  = label[i];
    const float sqi = G[(long)i * Bn + i];
    float ap = -__builtin_inff(), an = __builtin_inff();
    for (int j = tid; j < Bn; j += 256) {
        float d2   = sqi + G[(long)j * Bn + j] - 2.0f * G[(long)i * Bn + j];
        float dist = sqrtf(fmaxf(d2, 1e-12f));
        if (label[j] == li) { if (dist > ap) ap = dist; }
        else                { if (dist < an) an = dist; }
    }
    red[tid] = ap; __syncthreads();
    for (int k = 128; k > 0; k >>= 1) { if (tid < k) red[tid] = fmaxf(red[tid], red[tid + k]); __syncthreads(); }
    ap = red[0]; __syncthreads();
    red[tid] = an; __syncthreads();
    for (int k = 128; k > 0; k >>= 1) { if (tid < k) red[tid] = fminf(red[tid], red[tid + k]); __syncthreads(); }
    an = red[0];
    if (tid == 0) atomicAdd(&acc[accIdx], fmaxf(ap - an + 0.3f, 0.0f));
}

__global__ void zero_kernel(float* p, int n) {
    int i = blockIdx.x * blockDim.x + threadIdx.x;
    if (i < n) p[i] = 0.0f;
}

__global__ void finalize_kernel(const float* __restrict__ acc, float* __restrict__ out) {
    if (threadIdx.x == 0) {
        const float invB = 1.0f / (float)cB;
        const float loss_teacher = (acc[ACC_CET] + acc[ACC_TRT]) * invB;
        const float loss_student = (acc[ACC_CES] + acc[ACC_TRS]) * invB;
        const float loss_global  = acc[ACC_KL] * invB;
        const float loss_local   = 1.0f - acc[ACC_SIMOP] * invB;
        out[0] = loss_local + loss_global + loss_teacher + loss_student;
    }
}

// ---------------------------------------------------------------------------
// Host-side orchestration
// ---------------------------------------------------------------------------
extern "C" void kernel_launch(void* const* d_in, const int* in_sizes, int n_in,
                              void* d_out, int out_size, void* d_ws, size_t ws_size,
                              hipStream_t stream) {
    (void)in_sizes; (void)n_in; (void)out_size; (void)ws_size;

    const float* text_feat          = (const float*)d_in[0];   // (B, 77, 768)
    const float* teacher_image_raw  = (const float*)d_in[1];   // (B, 768)
    const float* student_global_raw = (const float*)d_in[2];   // (B, 576)
    const float* student_local_feat = (const float*)d_in[3];   // (B, 196, 576)
    const int*   label              = (const int*)d_in[4];     // (B,)
    const float* W_tv = (const float*)d_in[5];  const float* b_tv = (const float*)d_in[6];
    const float* W_tt = (const float*)d_in[7];  const float* b_tt = (const float*)d_in[8];
    const float* W_sv = (const float*)d_in[9];  const float* b_sv = (const float*)d_in[10];
    const float* bn_g = (const float*)d_in[11]; const float* bn_b = (const float*)d_in[12];
    const float* W_ct = (const float*)d_in[13]; const float* b_ct = (const float*)d_in[14];
    const float* W_cs = (const float*)d_in[15]; const float* b_cs = (const float*)d_in[16];
    float* out = (float*)d_out;

    // workspace partition (256B-aligned slabs)
    auto align256 = [](size_t x) { return (x + 255) & ~(size_t)255; };
    char* w = (char*)d_ws;
    float*  acc      = (float*)w;  w += 256;
    float*  lin_t    = (float*)w;  w += align256((size_t)cB * cFEAT * 4);           // teacher feat (BN'd in place)
    float*  sg       = (float*)w;  w += align256((size_t)cB * cFEAT * 4);           // student global feat
    float*  logits_t = (float*)w;  w += align256((size_t)cB * cNCLS * 4);
    float*  logits_s = (float*)w;  w += align256((size_t)cB * cNCLS * 4);
    float*  gram_t   = (float*)w;  w += align256((size_t)cB * cB * 4);
    float*  gram_s   = (float*)w;  w += align256((size_t)cB * cB * 4);
    float*  tt       = (float*)w;                                                   // teacher text f32 (B*77*576)
    float*  sim      = tt;         w += align256((size_t)cB * cN * cLDIM * 4);      // sim reuses tt slab
    __bf16* tnorm    = (__bf16*)w; w += align256((size_t)cB * cN * cLDIM * 2);
    __bf16* snorm    = (__bf16*)w; w += align256((size_t)cB * cM * cLDIM * 2);

    // block tile = 128(M) x 64(N), wave tile = 32x32
    auto gdim = [](int M, int N, int batch) {
        return dim3((unsigned)((N + 63) / 64), (unsigned)((M + 127) / 128), (unsigned)batch);
    };
    const dim3 blk(256, 1, 1);

    // 0) zero accumulators
    zero_kernel<<<1, 32, 0, stream>>>(acc, 8);

    // 1) teacher image: lin_t = raw @ W_tv^T + b_tv ; BN in place   (aligned)
    gemm_tn_wmma<float, false><<<gdim(cB, cFEAT, 1), blk, 0, stream>>>(
        teacher_image_raw, W_tv, b_tv, lin_t, cB, cFEAT, cTDIM, 0, 0, 0);
    bn_train_kernel<<<cFEAT, blk, 0, stream>>>(lin_t, bn_g, bn_b, cB, cFEAT);

    // 2) student global: sg = raw @ W_sv^T + b_sv                    (aligned)
    gemm_tn_wmma<float, false><<<gdim(cB, cFEAT, 1), blk, 0, stream>>>(
        student_global_raw, W_sv, b_sv, sg, cB, cFEAT, cLDIM, 0, 0, 0);

    // 3) classifier logits (N = 632 -> edge tiles)
    gemm_tn_wmma<float, true><<<gdim(cB, cNCLS, 1), blk, 0, stream>>>(
        lin_t, W_ct, b_ct, logits_t, cB, cNCLS, cFEAT, 0, 0, 0);
    gemm_tn_wmma<float, true><<<gdim(cB, cNCLS, 1), blk, 0, stream>>>(
        sg, W_cs, b_cs, logits_s, cB, cNCLS, cFEAT, 0, 0, 0);

    // 4) Gram matrices for triplet mining                            (aligned)
    gemm_tn_wmma<float, false><<<gdim(cB, cB, 1), blk, 0, stream>>>(
        lin_t, lin_t, nullptr, gram_t, cB, cB, cFEAT, 0, 0, 0);
    gemm_tn_wmma<float, false><<<gdim(cB, cB, 1), blk, 0, stream>>>(
        sg, sg, nullptr, gram_s, cB, cB, cFEAT, 0, 0, 0);

    // 5) teacher text projection: (B*77, 768) @ (576, 768)^T — big GEMM (aligned)
    gemm_tn_wmma<float, false><<<gdim(cB * cN, cLDIM, 1), blk, 0, stream>>>(
        text_feat, W_tt, b_tt, tt, cB * cN, cLDIM, cTDIM, 0, 0, 0);

    // 6) L2-normalize rows -> bf16 (tt slab is then reused for sim)
    rownorm_bf16_kernel<<<cB * cN, blk, 0, stream>>>(tt, tnorm, cLDIM);
    rownorm_bf16_kernel<<<cB * cM, blk, 0, stream>>>(student_local_feat, snorm, cLDIM);

    // 7) batched sim = snorm @ tnorm^T : 512 x (196x576)(576x77)     (edges)
    gemm_tn_wmma<__bf16, true><<<gdim(cM, cN, cB), blk, 0, stream>>>(
        snorm, tnorm, nullptr, sim, cM, cN, cLDIM,
        (long)cM * cLDIM, (long)cN * cLDIM, (long)cM * cN);

    // 8) Sinkhorn + sum(T*sim) accumulation
    sinkhorn_kernel<<<cB, blk, 0, stream>>>(sim, acc);

    // 9) CE + KL + preds
    ce_kl_kernel<<<cB, blk, 0, stream>>>(logits_t, logits_s, label, acc, out + 1);

    // 10) triplet losses
    triplet_kernel<<<cB, blk, 0, stream>>>(gram_t, label, acc, ACC_TRT);
    triplet_kernel<<<cB, blk, 0, stream>>>(gram_s, label, acc, ACC_TRS);

    // 11) combine
    finalize_kernel<<<1, 32, 0, stream>>>(acc, out);
}
